// TernaryLinear_43576738185533
// MI455X (gfx1250) — matrix operand order repaired and checked by reading, not compiled, not run on hardware
//
#include <hip/hip_runtime.h>

// ---------------------------------------------------------------------------
// TernaryLinear for MI455X (gfx1250): out = (x @ ternary(W).T + bias) * scale
//   Pass 1: ternarize W -> f16 {-1,0,+1}; convert x -> f16 (workspace)
//   Pass 2: LDS double-buffered WMMA GEMM:
//           GLOBAL_LOAD_ASYNC_TO_LDS_B128 staging (ASYNCcnt), ds_load_b128
//           fragment reads, v_wmma_f32_16x16x32_f16 compute, fused epilogue.
// ---------------------------------------------------------------------------

typedef __attribute__((ext_vector_type(16))) _Float16 v16h;
typedef __attribute__((ext_vector_type(8)))  _Float16 v8h;
typedef __attribute__((ext_vector_type(4)))  _Float16 v4h;
typedef __attribute__((ext_vector_type(8)))  float    v8f;
typedef __attribute__((ext_vector_type(4)))  float    v4f;

union FragH { v16h f; v8h h[2]; };

#define AS1 __attribute__((address_space(1)))
#define AS3 __attribute__((address_space(3)))

// Matches the builtin's parameter type reported by clang:
// '__attribute__((__vector_size__(4 * sizeof(int)))) int *'
typedef int v4i_vs __attribute__((vector_size(16)));

// 16B global -> LDS async copy (per-lane), tracked by ASYNCcnt.
__device__ __forceinline__ void async_copy_b128(const _Float16* g, _Float16* l) {
#if __has_builtin(__builtin_amdgcn_global_load_async_to_lds_b128)
  __builtin_amdgcn_global_load_async_to_lds_b128(
      (AS1 v4i_vs*)g, (AS3 v4i_vs*)l, /*offset=*/0, /*cpol=*/0);
#else
  unsigned loff = (unsigned)(__SIZE_TYPE__)(AS3 void*)l;
  asm volatile("global_load_async_to_lds_b128 %0, %1, off"
               :: "v"(loff), "v"(g) : "memory");
#endif
}

__device__ __forceinline__ void wait_async0() {
#if __has_builtin(__builtin_amdgcn_s_wait_asynccnt)
  __builtin_amdgcn_s_wait_asynccnt(0);
#else
  asm volatile("s_wait_asynccnt 0x0" ::: "memory");
#endif
}

// ---- Pass 1a: ternarize weight, f32 -> f16 ---------------------------------
__global__ __launch_bounds__(256) void k_ternarize(const float* __restrict__ w,
                                                   _Float16* __restrict__ wh,
                                                   int n4) {
  int i = blockIdx.x * blockDim.x + threadIdx.x;
  if (i >= n4) return;
  v4f v = reinterpret_cast<const v4f*>(w)[i];
  v4h o;
#pragma unroll
  for (int j = 0; j < 4; ++j) {
    float t = (__builtin_fabsf(v[j]) < 0.1f) ? 0.0f : (v[j] > 0.0f ? 1.0f : -1.0f);
    o[j] = (_Float16)t;
  }
  reinterpret_cast<v4h*>(wh)[i] = o;
}

// ---- Pass 1b: convert activations, f32 -> f16 ------------------------------
__global__ __launch_bounds__(256) void k_cvt_f16(const float* __restrict__ x,
                                                 _Float16* __restrict__ xh,
                                                 int n4) {
  int i = blockIdx.x * blockDim.x + threadIdx.x;
  if (i >= n4) return;
  v4f v = reinterpret_cast<const v4f*>(x)[i];
  v4h o;
#pragma unroll
  for (int j = 0; j < 4; ++j) o[j] = (_Float16)v[j];
  reinterpret_cast<v4h*>(xh)[i] = o;
}

// ---- Pass 2: LDS double-buffered WMMA GEMM ---------------------------------
// Block: 128 threads = 4 waves in a 2x2 grid; wave tile 64x64 (4x4 frags).
// Block tile 128x128, K staged 32 wide, double buffered in LDS.
#define BM 128
#define BN 128
#define BK 32
#define LDSW 40   // halves per LDS row: 32 data + 8 pad (80B, 16-bank skew)

__global__ __launch_bounds__(128) void k_gemm_wmma(
    const _Float16* __restrict__ X,   // [M,K] f16
    const _Float16* __restrict__ W,   // [N,K] f16 ternary
    const float* __restrict__ scale,  // [N]
    const float* __restrict__ bias,   // [N]
    float* __restrict__ out,          // [M,N]
    int M, int N, int K) {
  __shared__ __attribute__((aligned(16))) _Float16 ldsA[2][BM * LDSW];
  __shared__ __attribute__((aligned(16))) _Float16 ldsB[2][BN * LDSW];

  const int t    = threadIdx.x;        // 0..127
  const int lane = t & 31;
  const int wave = t >> 5;             // 0..3
  const int wm   = wave >> 1;          // 0..1
  const int wn   = wave & 1;           // 0..1

  const int blockM = blockIdx.y * BM;
  const int blockN = blockIdx.x * BN;

  // Staging: thread t owns row t of both tiles (64B = 4 x b128 per tile).
  const _Float16* gA = X + (size_t)(blockM + t) * K;
  const _Float16* gB = W + (size_t)(blockN + t) * K;

  // Fragment geometry (v_wmma_f32_16x16x32_f16 VGPR layouts).
  const int lrow = lane & 15;          // A row / B out-channel within frag
  const int g    = lane >> 4;          // lane group
  const int kbA  = g * 8;              // A: halves 0-7 @K=kbA, 8-15 @K=16+kbA
  const int kbB  = g * 8;              // B: halves 0-7 @K=kbB, 8-15 @K=16+kbB

  v8f acc[4][4] = {};

  // Prologue: stage k=0 into buffer 0.
#pragma unroll
  for (int q = 0; q < 4; ++q) {
    async_copy_b128(gA + q * 8, &ldsA[0][t * LDSW + q * 8]);
    async_copy_b128(gB + q * 8, &ldsB[0][t * LDSW + q * 8]);
  }

  const int KT = K / BK;
  for (int kt = 0; kt < KT; ++kt) {
    const int buf = kt & 1;
    wait_async0();        // this wave's stage of `buf` has landed
    __syncthreads();      // every wave's stage of `buf` has landed

    if (kt + 1 < KT) {    // prefetch next K-slab into the other buffer
      const int kn = (kt + 1) * BK;
#pragma unroll
      for (int q = 0; q < 4; ++q) {
        async_copy_b128(gA + kn + q * 8, &ldsA[buf ^ 1][t * LDSW + q * 8]);
        async_copy_b128(gB + kn + q * 8, &ldsB[buf ^ 1][t * LDSW + q * 8]);
      }
    }

    FragH a[4], b[4];
#pragma unroll
    for (int i = 0; i < 4; ++i) {
      const _Float16* p = &ldsA[buf][(wm * 64 + i * 16 + lrow) * LDSW];
      a[i].h[0] = *reinterpret_cast<const v8h*>(p + kbA);
      a[i].h[1] = *reinterpret_cast<const v8h*>(p + kbA + 16);
    }
#pragma unroll
    for (int j = 0; j < 4; ++j) {
      const _Float16* p = &ldsB[buf][(wn * 64 + j * 16 + lrow) * LDSW];
      b[j].h[0] = *reinterpret_cast<const v8h*>(p + kbB);
      b[j].h[1] = *reinterpret_cast<const v8h*>(p + kbB + 16);
    }
#pragma unroll
    for (int i = 0; i < 4; ++i)
#pragma unroll
      for (int j = 0; j < 4; ++j)
        acc[i][j] = __builtin_amdgcn_wmma_f32_16x16x32_f16(
            false, a[i].f, false, b[j].f, (short)0, acc[i][j], false, false);
  }

  // Epilogue: lane holds N = lane&15, M = g*8 + v within each 16x16 frag.
  const int ncol = lane & 15;
  const int moff = g * 8;
#pragma unroll
  for (int j = 0; j < 4; ++j) {
    const int col = blockN + wn * 64 + j * 16 + ncol;
    const float s = scale[col];
    const float bsum = bias[col];
#pragma unroll
    for (int i = 0; i < 4; ++i) {
#pragma unroll
      for (int v = 0; v < 8; ++v) {
        const int r = blockM + wm * 64 + i * 16 + moff + v;
        out[(size_t)r * N + col] = (acc[i][j][v] + bsum) * s;
      }
    }
  }
}

// ---------------------------------------------------------------------------
extern "C" void kernel_launch(void* const* d_in, const int* in_sizes, int n_in,
                              void* d_out, int out_size, void* d_ws, size_t ws_size,
                              hipStream_t stream) {
  const float* x     = (const float*)d_in[0];
  const float* w     = (const float*)d_in[1];
  const float* scale = (const float*)d_in[2];
  const float* bias  = (const float*)d_in[3];

  const int N = in_sizes[2];           // OUT channels
  const int K = in_sizes[1] / N;       // IN features
  const int M = in_sizes[0] / K;       // batch rows

  _Float16* Xh = reinterpret_cast<_Float16*>(d_ws);
  _Float16* Wh = reinterpret_cast<_Float16*>(
      reinterpret_cast<char*>(d_ws) + (size_t)M * K * sizeof(_Float16));

  const int xn4 = (M * K) / 4;
  const int wn4 = (N * K) / 4;
  k_cvt_f16<<<(xn4 + 255) / 256, 256, 0, stream>>>(x, Xh, xn4);
  k_ternarize<<<(wn4 + 255) / 256, 256, 0, stream>>>(w, Wh, wn4);

  dim3 grid(N / BN, M / BM);
  k_gemm_wmma<<<grid, 128, 0, stream>>>(Xh, Wh, scale, bias,
                                        (float*)d_out, M, N, K);
}
